// MultiHeadAttentionBlock_54674933678846
// MI455X (gfx1250) — matrix operand order repaired and checked by reading, not compiled, not run on hardware
//
#include <hip/hip_runtime.h>
#include <hip/hip_bf16.h>
#include <stdint.h>

// Problem constants (B=2, T=1024, F=1024, H=16, D=64, CTX=1024, S=2048)
#define NB   2
#define NT   1024
#define NF   1024
#define NH   16
#define ND   64
#define NCTX 1024
#define NS   2048
#define NM   (NB * NT)          // 2048 token rows
#define BIG_NEG (-0.7f * 3.402823466e38f)

// LDS tile geometry for the GEMMs: 64(M) x 128(N) block tile, K-step 64,
// row pitch 72 bf16 (144B = 36 banks; 36*r mod 64 distinct for r=0..15 ->
// conflict-free b128 fragment reads).
#define KSTEP   64
#define APITCH  72
#define AROWS   64
#define BROWS   128
#define ABUF_E  (AROWS * APITCH)   // elems per A buffer (4608)
#define BBUF_E  (BROWS * APITCH)   // elems per B buffer (9216)

typedef __attribute__((ext_vector_type(16))) __bf16 v16bf;
typedef __attribute__((ext_vector_type(8)))  __bf16 v8bf;
typedef __attribute__((ext_vector_type(8)))  float  v8f;

static __device__ __forceinline__ v8f wmma_bf16(v16bf a, v16bf b, v8f c) {
  // D = A(16x32 bf16) * B(32x16 bf16) + C(16x16 f32)
  return __builtin_amdgcn_wmma_f32_16x16x32_bf16(false, a, false, b, (short)0, c,
                                                 false, false);
}

// A-operand fragment: per lane, 16 contiguous bf16 at p
static __device__ __forceinline__ v16bf load_frag_a(const __bf16* p) {
  v8bf lo = *(const v8bf*)p;
  v8bf hi = *(const v8bf*)(p + 8);
  return __builtin_shufflevector(lo, hi, 0, 1, 2, 3, 4, 5, 6, 7, 8, 9, 10, 11,
                                 12, 13, 14, 15);
}
// B-operand fragment: per lane, 16 contiguous bf16 at p
static __device__ __forceinline__ v16bf load_frag_b(const __bf16* p) {
  v8bf lo = *(const v8bf*)p;
  v8bf hi = *(const v8bf*)(p + 8);
  return __builtin_shufflevector(lo, hi, 0, 1, 2, 3, 4, 5, 6, 7, 8, 9, 10, 11,
                                 12, 13, 14, 15);
}

// CDNA5 async copy: 16 bytes per lane, global -> LDS, tracked by ASYNCcnt.
static __device__ __forceinline__ void async_copy_b128(uint32_t lds_off,
                                                       const __bf16* g) {
  asm volatile("global_load_async_to_lds_b128 %0, %1, off"
               :
               : "v"(lds_off), "v"(g)
               : "memory");
}
static __device__ __forceinline__ void wait_async6() {
  asm volatile("s_wait_asynccnt 0x6" ::: "memory");
}
static __device__ __forceinline__ void wait_async0() {
  asm volatile("s_wait_asynccnt 0x0" ::: "memory");
}

// ---------------- shared GEMM core (double-buffered async-to-LDS) ----------
struct Acc4 {
  v8f c00, c01, c10, c11;
};

// Ag: block A base (row mblk of [*,1024] bf16, K-contiguous)
// Bg: block B base (row nblk of [*,1024] bf16, K-contiguous, pre-transposed)
// lds: 2*ABUF_E + 2*BBUF_E bf16 elems of shared memory
static __device__ __forceinline__ Acc4 gemm_block_lds(const __bf16* Ag,
                                                      const __bf16* Bg,
                                                      __bf16* lds) {
  const int tid = threadIdx.x;
  const int lane = tid & 31, wave = tid >> 5;
  const int l16 = lane & 15, lh = lane >> 4;
  const int wm = wave >> 2, wn = wave & 3;

  __bf16* As = lds;                // [2][ABUF_E]
  __bf16* Bs = lds + 2 * ABUF_E;   // [2][BBUF_E]

  // copy lane assignment: 256 lanes x 16B per async op; rows split j*32
  const int r0 = tid >> 3;         // 0..31
  const int k8 = (tid & 7) * 8;    // K elem offset 0..56
  const uint32_t as_base = (uint32_t)(uintptr_t)As;
  const uint32_t bs_base = (uint32_t)(uintptr_t)Bs;
  const uint32_t a_l = as_base + (uint32_t)(r0 * APITCH + k8) * 2;
  const uint32_t b_l = bs_base + (uint32_t)(r0 * APITCH + k8) * 2;
  const __bf16* ag = Ag + (size_t)r0 * NF + k8;
  const __bf16* bg = Bg + (size_t)r0 * NF + k8;
  const uint32_t ABYTES = ABUF_E * 2;
  const uint32_t BBYTES = BBUF_E * 2;
  const uint32_t ROWSTEP = 32 * APITCH * 2;  // 32 rows of LDS, bytes

  Acc4 acc;
  acc.c00 = {}; acc.c01 = {}; acc.c10 = {}; acc.c11 = {};

  auto issue = [&](int stage, int buf) {
    const int kof = stage * KSTEP;
    const uint32_t la = buf ? ABYTES : 0u;
    const uint32_t lb = buf ? BBYTES : 0u;
    async_copy_b128(a_l + la, ag + kof);
    async_copy_b128(a_l + la + ROWSTEP, ag + kof + (size_t)32 * NF);
    async_copy_b128(b_l + lb, bg + kof);
    async_copy_b128(b_l + lb + ROWSTEP, bg + kof + (size_t)32 * NF);
    async_copy_b128(b_l + lb + 2 * ROWSTEP, bg + kof + (size_t)64 * NF);
    async_copy_b128(b_l + lb + 3 * ROWSTEP, bg + kof + (size_t)96 * NF);
  };
  auto compute = [&](int buf) {
    const __bf16* Ap =
        As + (buf ? ABUF_E : 0) + (wm * 32 + l16) * APITCH + lh * 16;
    const __bf16* Bp =
        Bs + (buf ? BBUF_E : 0) + (wn * 32 + l16) * APITCH + lh * 16;
#pragma unroll
    for (int s = 0; s < 2; ++s) {
      const int ko = s * 32;
      v16bf A0 = load_frag_a(Ap + ko);
      v16bf A1 = load_frag_a(Ap + 16 * APITCH + ko);
      v16bf B0 = load_frag_b(Bp + ko);
      v16bf B1 = load_frag_b(Bp + 16 * APITCH + ko);
      acc.c00 = wmma_bf16(A0, B0, acc.c00);
      acc.c01 = wmma_bf16(A0, B1, acc.c01);
      acc.c10 = wmma_bf16(A1, B0, acc.c10);
      acc.c11 = wmma_bf16(A1, B1, acc.c11);
    }
  };

  // 16 stages of K=64, two-stage software pipeline, explicit even/odd buffers
  issue(0, 0);
#pragma unroll 1
  for (int kt = 0; kt < 14; kt += 2) {
    issue(kt + 1, 1);
    wait_async6();
    __syncthreads();
    compute(0);
    __syncthreads();
    issue(kt + 2, 0);
    wait_async6();
    __syncthreads();
    compute(1);
    __syncthreads();
  }
  issue(15, 1);
  wait_async6();
  __syncthreads();
  compute(0);
  __syncthreads();
  wait_async0();
  __syncthreads();
  compute(1);
  return acc;
}

// ---------------- elementwise conversion kernels ----------------

__global__ void mha_cvt_x(const float* __restrict__ x, __bf16* __restrict__ xb,
                          int n) {
  int i = blockIdx.x * blockDim.x + threadIdx.x;
  if (i < n) xb[i] = (__bf16)x[i];
}

// Wq/Wk/Wv: (F,H,D) -> WqkvT bf16 [3072][1024] (N-major, K=F contiguous)
// Wo: (H,D,F) -> WoT bf16 [1024(f)][1024(hd)]
__global__ void mha_cvt_w(const float* __restrict__ Wq,
                          const float* __restrict__ Wk,
                          const float* __restrict__ Wv,
                          const float* __restrict__ Wo,
                          __bf16* __restrict__ WqkvT,
                          __bf16* __restrict__ WoT) {
  int i = blockIdx.x * blockDim.x + threadIdx.x;  // i = f*1024 + hd
  if (i >= NF * NH * ND) return;
  int f = i >> 10, hd = i & 1023;
  WqkvT[(size_t)(hd)*NF + f]          = (__bf16)Wq[i];
  WqkvT[(size_t)(1024 + hd) * NF + f] = (__bf16)Wk[i];
  WqkvT[(size_t)(2048 + hd) * NF + f] = (__bf16)Wv[i];
  WoT[(size_t)f * 1024 + hd]          = (__bf16)Wo[(size_t)hd * NF + f];
}

__global__ void mha_cvt_bias(const float* __restrict__ bq,
                             const float* __restrict__ bk,
                             const float* __restrict__ bv,
                             float* __restrict__ cbias) {
  int i = blockIdx.x * blockDim.x + threadIdx.x;
  if (i >= 3 * 1024) return;
  int sec = i >> 10, hd = i & 1023;
  cbias[i] = (sec == 0) ? bq[hd] : (sec == 1) ? bk[hd] : bv[hd];
}

// cache_key/value: (B,CTX,H,D) f32 -> Kb[bh][s][d] bf16, VbT[bh][d][s] bf16
__global__ void mha_cvt_cache(const float* __restrict__ ck,
                              const float* __restrict__ cv,
                              __bf16* __restrict__ Kb,
                              __bf16* __restrict__ VbT) {
  int i = blockIdx.x * blockDim.x + threadIdx.x;
  if (i >= NB * NCTX * NH * ND) return;
  int d = i & 63, h = (i >> 6) & 15, s = (i >> 10) & 1023, b = i >> 20;
  int bh = b * NH + h;
  Kb[((size_t)bh * NS + s) * ND + d]  = (__bf16)ck[i];
  VbT[((size_t)bh * ND + d) * NS + s] = (__bf16)cv[i];
}

// reverse cumsum of [cache_mask, mask] per batch; also new_mask / new_position
__global__ void mha_scan(const uint8_t* __restrict__ mask,
                         const uint8_t* __restrict__ cmask,
                         const int* __restrict__ cpos,
                         int* __restrict__ kin,
                         int* __restrict__ out_mask,
                         int* __restrict__ out_pos) {
  int b = threadIdx.x;
  if (b >= NB) return;
  int run = 0;
  for (int s = NS - 1; s >= 0; --s) {
    int bit = (s >= NCTX) ? (mask[b * NT + s - NCTX] != 0)
                          : (cmask[b * NCTX + s] != 0);
    run += bit;
    kin[b * NS + s] = run;
    if (s >= NCTX) out_mask[b * NT + s - NCTX] = bit;
  }
  int qsum = kin[b * NS + NCTX];  // sum of new mask
  out_pos[b] = (cpos[b] + qsum) & NCTX;
}

// ---------------- QKV projection GEMM (async-LDS + WMMA bf16) --------------
// C[2048][3072] = xb[2048][1024] * WqkvT^T + cbias ; scatter into Q/K/V dests
__global__ void __launch_bounds__(256) mha_gemm_qkv(
    const __bf16* __restrict__ xb, const __bf16* __restrict__ WT,
    const float* __restrict__ cbias, __bf16* __restrict__ Qb,
    float* __restrict__ nK, __bf16* __restrict__ Kb, float* __restrict__ nV,
    __bf16* __restrict__ VbT) {
  __shared__ __bf16 lds[2 * ABUF_E + 2 * BBUF_E];
  const int lane = threadIdx.x & 31;
  const int wave = threadIdx.x >> 5;
  const int l16 = lane & 15, lh = lane >> 4;
  const int mblk = blockIdx.x * 64;
  const int nblk = blockIdx.y * 128;
  const int mtile = mblk + (wave >> 2) * 32;
  const int ntile = nblk + (wave & 3) * 32;

  Acc4 acc = gemm_block_lds(xb + (size_t)mblk * NF, WT + (size_t)nblk * NF, lds);

  const float bias0 = cbias[ntile + l16];
  const float bias1 = cbias[ntile + 16 + l16];

#pragma unroll
  for (int mi = 0; mi < 2; ++mi) {
#pragma unroll
    for (int ni = 0; ni < 2; ++ni) {
      v8f a = (mi == 0) ? ((ni == 0) ? acc.c00 : acc.c01)
                        : ((ni == 0) ? acc.c10 : acc.c11);
      float bias = (ni == 0) ? bias0 : bias1;
      int gn = ntile + ni * 16 + l16;
      int sec = gn >> 10, hd = gn & 1023, h = hd >> 6, d = hd & 63;
#pragma unroll
      for (int r = 0; r < 8; ++r) {
        int gm = mtile + mi * 16 + r + 8 * lh;
        int bb = gm >> 10, t = gm & 1023;
        float val = a[r] + bias;
        int bh = bb * NH + h;
        if (sec == 0) {
          Qb[((size_t)bh * NT + t) * ND + d] = (__bf16)val;
        } else if (sec == 1) {
          nK[(((size_t)(bb * NT + t)) * NH + h) * ND + d] = val;
          Kb[((size_t)bh * NS + NCTX + t) * ND + d] = (__bf16)val;
        } else {
          nV[(((size_t)(bb * NT + t)) * NH + h) * ND + d] = val;
          VbT[((size_t)bh * ND + d) * NS + NCTX + t] = (__bf16)val;
        }
      }
    }
  }
}

// ---------------- attention (flash-style, logits^T trick) ----------------
__global__ void __launch_bounds__(256) mha_attn(
    const __bf16* __restrict__ Qb, const __bf16* __restrict__ Kb,
    const __bf16* __restrict__ VbT, const int* __restrict__ kin,
    __bf16* __restrict__ attnb) {
  const int lane = threadIdx.x & 31, wave = threadIdx.x >> 5;
  const int l16 = lane & 15, lh = lane >> 4;
  const int bh = blockIdx.x >> 3;     // 32 (b,h) pairs
  const int b = bh >> 4, h = bh & 15;
  const int tbase = (blockIdx.x & 7) * 128 + wave * 16;
  const int t = tbase + l16;
  const float scale = 0.125f;  // 1/sqrt(64)

  const __bf16* q = Qb + ((size_t)bh * NT + t) * ND;
  v16bf Bq0 = load_frag_b(q + lh * 16);       // d 0..31
  v16bf Bq1 = load_frag_b(q + 32 + lh * 16);  // d 32..63
  const int qin = kin[b * NS + NCTX + t];
  const int* kinb = kin + b * NS;
  const __bf16* kb = Kb + (size_t)bh * NS * ND;
  const __bf16* vtb = VbT + (size_t)bh * ND * NS;

  float m = BIG_NEG, l = 0.f;
  v8f o0 = {}, o1 = {}, o2 = {}, o3 = {};
  const int send = tbase + 16;  // causal: s <= t
  for (int sb = 0; sb < send; sb += 32) {
    // logits^T tiles: c0 -> s in [sb,sb+16), c1 -> s in [sb+16,sb+32)
    const __bf16* kr0 = kb + (size_t)(sb + l16) * ND + lh * 16;
    const __bf16* kr1 = kr0 + 16 * ND;
    // prefetch next key/value tiles into cache (global_prefetch_b8)
    __builtin_prefetch(kr0 + 32 * ND, 0, 1);
    __builtin_prefetch(vtb + (size_t)l16 * NS + sb + 32, 0, 1);
    v8f c0 = {}, c1 = {};
    c0 = wmma_bf16(load_frag_a(kr0), Bq0, c0);
    c0 = wmma_bf16(load_frag_a(kr0 + 32), Bq1, c0);
    c1 = wmma_bf16(load_frag_a(kr1), Bq0, c1);
    c1 = wmma_bf16(load_frag_a(kr1 + 32), Bq1, c1);

    float p0[8], p1[8];
    float mx = m;
#pragma unroll
    for (int r = 0; r < 8; ++r) {
      int s0 = sb + 8 * lh + r;
      int s1 = s0 + 16;
      float v0 = (s0 <= t && kinb[s0] == qin) ? c0[r] * scale : BIG_NEG;
      float v1 = (s1 <= t && kinb[s1] == qin) ? c1[r] * scale : BIG_NEG;
      p0[r] = v0;
      p1[r] = v1;
      mx = fmaxf(mx, fmaxf(v0, v1));
    }
    mx = fmaxf(mx, __shfl_xor(mx, 16, 32));  // merge half-lanes of same t
    float alpha = __expf(m - mx);
    float lsum = 0.f;
#pragma unroll
    for (int r = 0; r < 8; ++r) {
      p0[r] = __expf(p0[r] - mx);
      p1[r] = __expf(p1[r] - mx);
      lsum += p0[r] + p1[r];
    }
    lsum += __shfl_xor(lsum, 16, 32);
    l = l * alpha + lsum;
    m = mx;
#pragma unroll
    for (int r = 0; r < 8; ++r) {
      o0[r] *= alpha; o1[r] *= alpha; o2[r] *= alpha; o3[r] *= alpha;
    }
    // Build B-operand of P (K = s_rel 0..31, N = t) via half-lane exchange
    v16bf P;
#pragma unroll
    for (int r = 0; r < 8; ++r) {
      float q0 = __shfl_xor(p0[r], 16, 32);
      float q1 = __shfl_xor(p1[r], 16, 32);
      float lo = (lh == 0) ? p0[r] : q1;
      float hi = (lh == 0) ? q0 : p1[r];
      P[r] = (__bf16)lo;
      P[r + 8] = (__bf16)hi;
    }
    // O^T += V^T * P  (A = V^T rows d, K = s)
    const size_t vstep = (size_t)16 * NS;
    const __bf16* vr = vtb + (size_t)l16 * NS + sb + lh * 16;
    o0 = wmma_bf16(load_frag_a(vr), P, o0);
    o1 = wmma_bf16(load_frag_a(vr + vstep), P, o1);
    o2 = wmma_bf16(load_frag_a(vr + 2 * vstep), P, o2);
    o3 = wmma_bf16(load_frag_a(vr + 3 * vstep), P, o3);
  }
  const float inv = 1.f / l;
  __bf16* outp = attnb + ((size_t)(b * NT + t)) * (NH * ND) + h * ND + lh * 8;
#define STORE_O(of, foff)                                   \
  {                                                         \
    v8bf pk;                                                \
    _Pragma("unroll") for (int r = 0; r < 8; ++r)           \
        pk[r] = (__bf16)(of[r] * inv);                      \
    *(v8bf*)(outp + (foff)) = pk;                           \
  }
  STORE_O(o0, 0)
  STORE_O(o1, 16)
  STORE_O(o2, 32)
  STORE_O(o3, 48)
#undef STORE_O
}

// ---------------- output projection GEMM ----------------
// y[2048][1024] = attnb[2048][1024] * WoT^T + bo
__global__ void __launch_bounds__(256) mha_gemm_out(
    const __bf16* __restrict__ attnb, const __bf16* __restrict__ WoT,
    const float* __restrict__ bo, float* __restrict__ y) {
  __shared__ __bf16 lds[2 * ABUF_E + 2 * BBUF_E];
  const int lane = threadIdx.x & 31;
  const int wave = threadIdx.x >> 5;
  const int l16 = lane & 15, lh = lane >> 4;
  const int mblk = blockIdx.x * 64;
  const int nblk = blockIdx.y * 128;
  const int mtile = mblk + (wave >> 2) * 32;
  const int ntile = nblk + (wave & 3) * 32;

  Acc4 acc =
      gemm_block_lds(attnb + (size_t)mblk * 1024, WoT + (size_t)nblk * 1024, lds);

  const float bias0 = bo[ntile + l16];
  const float bias1 = bo[ntile + 16 + l16];
#pragma unroll
  for (int mi = 0; mi < 2; ++mi) {
#pragma unroll
    for (int ni = 0; ni < 2; ++ni) {
      v8f a = (mi == 0) ? ((ni == 0) ? acc.c00 : acc.c01)
                        : ((ni == 0) ? acc.c10 : acc.c11);
      float bias = (ni == 0) ? bias0 : bias1;
      int gn = ntile + ni * 16 + l16;
#pragma unroll
      for (int r = 0; r < 8; ++r) {
        int gm = mtile + mi * 16 + r + 8 * lh;
        y[(size_t)gm * NF + gn] = a[r] + bias;
      }
    }
  }
}

// ---------------- host-side orchestration ----------------

extern "C" void kernel_launch(void* const* d_in, const int* in_sizes, int n_in,
                              void* d_out, int out_size, void* d_ws,
                              size_t ws_size, hipStream_t stream) {
  (void)in_sizes; (void)n_in; (void)out_size; (void)ws_size;
  const float*   x     = (const float*)d_in[0];
  const uint8_t* mask  = (const uint8_t*)d_in[1];
  const float*   ck    = (const float*)d_in[2];
  const float*   cv    = (const float*)d_in[3];
  const uint8_t* cmask = (const uint8_t*)d_in[4];
  const int*     cpos  = (const int*)d_in[5];
  const float*   Wq    = (const float*)d_in[6];
  const float*   bq    = (const float*)d_in[7];
  const float*   Wk    = (const float*)d_in[8];
  const float*   bk    = (const float*)d_in[9];
  const float*   Wv    = (const float*)d_in[10];
  const float*   bv    = (const float*)d_in[11];
  const float*   Wo    = (const float*)d_in[12];
  const float*   bo    = (const float*)d_in[13];

  // workspace carve-out (256B aligned regions)
  char* w = (char*)d_ws;
  auto carve = [&](size_t bytes) {
    char* p = w;
    w += (bytes + 255) & ~(size_t)255;
    return p;
  };
  __bf16* xb    = (__bf16*)carve((size_t)NM * NF * 2);           // 4 MiB
  __bf16* WqkvT = (__bf16*)carve((size_t)3 * 1024 * NF * 2);     // 6 MiB
  __bf16* WoT   = (__bf16*)carve((size_t)NF * 1024 * 2);         // 2 MiB
  float*  cbias = (float*)carve((size_t)3 * 1024 * 4);
  __bf16* Qb    = (__bf16*)carve((size_t)NB * NH * NT * ND * 2); // 4 MiB
  __bf16* Kb    = (__bf16*)carve((size_t)NB * NH * NS * ND * 2); // 8 MiB
  __bf16* VbT   = (__bf16*)carve((size_t)NB * NH * ND * NS * 2); // 8 MiB
  __bf16* attnb = (__bf16*)carve((size_t)NM * NH * ND * 2);      // 4 MiB
  int*    kin   = (int*)carve((size_t)NB * NS * 4);

  // output carve-out (flat, return order)
  float* y   = (float*)d_out;                      // (B,T,F)
  float* nK  = y + (size_t)NB * NT * NF;           // (B,CTX,H,D)
  float* nV  = nK + (size_t)NB * NCTX * NH * ND;   // (B,CTX,H,D)
  int*   nMk = (int*)(nV + (size_t)NB * NCTX * NH * ND);  // (B,CTX) int32
  int*   nPs = nMk + (size_t)NB * NCTX;            // (B,1) int32

  const int TPB = 256;
  int n;
  n = NM * NF;
  mha_cvt_x<<<(n + TPB - 1) / TPB, TPB, 0, stream>>>(x, xb, n);
  n = NF * NH * ND;
  mha_cvt_w<<<(n + TPB - 1) / TPB, TPB, 0, stream>>>(Wq, Wk, Wv, Wo, WqkvT, WoT);
  n = 3 * 1024;
  mha_cvt_bias<<<(n + TPB - 1) / TPB, TPB, 0, stream>>>(bq, bk, bv, cbias);
  n = NB * NCTX * NH * ND;
  mha_cvt_cache<<<(n + TPB - 1) / TPB, TPB, 0, stream>>>(ck, cv, Kb, VbT);
  mha_scan<<<1, 32, 0, stream>>>(mask, cmask, cpos, kin, nMk, nPs);

  // QKV projection: M=2048 (64/block), N=3072 (128/block)
  mha_gemm_qkv<<<dim3(NM / 64, 3072 / 128), TPB, 0, stream>>>(
      xb, WqkvT, cbias, Qb, nK, Kb, nV, VbT);

  // attention: one wave per 16 t-rows; 8 waves/block -> 128 rows/block
  mha_attn<<<NB * NH * (NT / 128), TPB, 0, stream>>>(Qb, Kb, VbT, kin, attnb);

  // output projection: M=2048, N=1024
  mha_gemm_out<<<dim3(NM / 64, NF / 128), TPB, 0, stream>>>(attnb, WoT, bo, y);
}
